// DecoderAttn_66554813218820
// MI455X (gfx1250) — matrix operand order repaired and checked by reading, not compiled, not run on hardware
//
#include <hip/hip_runtime.h>
#include <hip/hip_fp16.h>

// ============================================================================
// DecoderAttn for MI455X (gfx1250, wave32, WMMA).
// - All matrix-shaped work (qkv/wo/mlp GEMMs, causal dilated convs, conv-
//   transpose phases, 1x1 convs) runs through one tapped WMMA GEMM kernel
//   using v_wmma_f32_16x16x32_f16 (f16 operands, f32 accumulate).
// - Local attention is fused flash-style (online softmax, WMMA for QK^T and
//   P*V); V chunks are staged into LDS with GLOBAL_LOAD_ASYNC_TO_LDS_B128
//   (ASYNCcnt-tracked DMA) per CDNA5 ISA.
// - Out-of-range conv tap rows read from a zeroed dummy row instead of being
//   EXEC-masked, keeping the GEMM inner loop branch-free.
// Workspace requirement ~300MB.
// ============================================================================

typedef _Float16 h16;
typedef __attribute__((ext_vector_type(16))) _Float16 v16h;
typedef __attribute__((ext_vector_type(8)))  float    v8f;

__device__ inline v8f v8zero() {
  v8f z = {0.f,0.f,0.f,0.f,0.f,0.f,0.f,0.f};
  return z;
}

__device__ inline v8f wmma16(v16h a, v16h b, v8f c) {
  // (neg_a, A, neg_b, B, c_mod, C, reuse_a, reuse_b)
  return __builtin_amdgcn_wmma_f32_16x16x32_f16(false, a, false, b, (short)0, c,
                                                false, false);
}

union V16U { v16h v; float4 q[2]; h16 e[16]; };

// B operand (K=32 x N=16): lane<16 -> n=lane, k=0..15 ; lane>=16 -> k=16..31.
// Caller bakes the (hi?16:0) k-offset into p. 16 contiguous halves.
__device__ inline v16h ld_b16(const h16* p) {
  V16U u;
  u.q[0] = *(const float4*)(p);
  u.q[1] = *(const float4*)(p + 8);
  return u.v;
}

// A operand (M=16 x K=32): lane<16 -> m=lane, elems {k0..k0+15};
// lane>=16 -> {k0+16..k0+31}. row = per-lane row pointer.
__device__ inline v16h ld_a16(const h16* row, int k0, bool hi) {
  V16U u;
  u.q[0] = *(const float4*)(row + k0 + (hi ? 16 : 0));
  u.q[1] = *(const float4*)(row + k0 + 8 + (hi ? 16 : 0));
  return u.v;
}

// ---------------------------------------------------------------------------
// Tapped WMMA GEMM:
//   Y[bz, row_out, col] = act( bias[col] + sum_t A[bz, row + (t-(ntaps-1))*dr]
//                                          @ Bt[w0 + t*wstep] ) (+ residual)
//   row_out = row*ostride + ophase.  Bt slice t is (Cout x Cin) f16 (B^T).
// Out-of-range tap rows are redirected to Z (a zeroed row of >=Cin halves),
// so the k-loop is branch-free: pure b128 clauses + 4 WMMAs.
// Block = 128 threads (4 waves), each wave owns a 32x32 output tile.
// ---------------------------------------------------------------------------
__global__ __launch_bounds__(128) void k_gemm_taps(
    const h16* __restrict__ A, const h16* __restrict__ W,
    const float* __restrict__ bias,
    const h16* __restrict__ R16, const float* __restrict__ R32,
    h16* __restrict__ Y16, float* __restrict__ Y32,
    const h16* __restrict__ Z,
    int M, int LA, int Cin, int Cout,
    int ntaps, int dr, int w0, int wstep,
    int ostride, int ophase,
    long long sAB, long long sYB, int act)
{
  const int wid  = threadIdx.x >> 5;
  const int lane = threadIdx.x & 31;
  const bool hi  = lane >= 16;
  const int ln   = lane & 15;
  const int mb = blockIdx.y * 64 + (wid >> 1) * 32;
  const int nb = blockIdx.x * 64 + (wid & 1) * 32;
  if (nb >= Cout) return;                     // wave-uniform exit
  const h16* Ab = A + (size_t)blockIdx.z * sAB;

  v8f c00 = v8zero(), c01 = v8zero(), c10 = v8zero(), c11 = v8zero();

  for (int t = 0; t < ntaps; ++t) {
    const int roff = (t - (ntaps - 1)) * dr;
    const h16* Wt = W + (size_t)(w0 + t * wstep) * (size_t)Cout * Cin;
    const int r0 = mb + ln + roff;
    const int r1 = r0 + 16;
    // zero-row redirect instead of EXEC-masked loads (one cndmask per tap)
    const h16* ar0 = ((r0 >= 0) && (r0 < LA)) ? Ab + (long long)r0 * Cin : Z;
    const h16* ar1 = ((r1 >= 0) && (r1 < LA)) ? Ab + (long long)r1 * Cin : Z;
    const h16* br0 = Wt + (size_t)(nb + ln) * Cin + (hi ? 16 : 0);
    const h16* br1 = Wt + (size_t)(nb + 16 + ln) * Cin + (hi ? 16 : 0);
    if (t + 1 < ntaps) {  // prefetch next tap's A rows (global_prefetch_b8)
      const int rn = r0 + dr;
      if (rn >= 0 && rn < LA) __builtin_prefetch(Ab + (long long)rn * Cin, 0, 0);
    }
    for (int k = 0; k < Cin; k += 32) {
      v16h a0 = ld_a16(ar0, k, hi);
      v16h a1 = ld_a16(ar1, k, hi);
      v16h b0 = ld_b16(br0 + k);
      v16h b1 = ld_b16(br1 + k);
      c00 = wmma16(a0, b0, c00);
      c01 = wmma16(a0, b1, c01);
      c10 = wmma16(a1, b0, c10);
      c11 = wmma16(a1, b1, c11);
    }
  }

  // Epilogue. C layout: VGPR r, lane l -> row = r + 8*(l>=16), col = l%16.
#pragma unroll
  for (int mt = 0; mt < 2; ++mt)
#pragma unroll
    for (int nt = 0; nt < 2; ++nt) {
      v8f cc = (mt == 0) ? (nt == 0 ? c00 : c01) : (nt == 0 ? c10 : c11);
#pragma unroll
      for (int r = 0; r < 8; ++r) {
        const int row = mb + mt * 16 + r + (hi ? 8 : 0);
        const int col = nb + nt * 16 + ln;
        if (row < M && col < Cout) {
          float v = cc[r];
          if (bias) v += bias[col];
          if (act == 1)       v = 0.5f * v * (1.f + erff(v * 0.70710678118f)); // exact gelu
          else if (act == 2)  v = (v > 0.f) ? v : expm1f(v);                   // elu
          const size_t oi = (size_t)blockIdx.z * sYB +
                            ((size_t)row * ostride + ophase) * (size_t)Cout + col;
          if (R32) v += R32[oi];
          if (R16) v += (float)R16[oi];
          if (Y32) Y32[oi] = v;
          if (Y16) Y16[oi] = (h16)v;
        }
      }
    }
}

// ---------------------------------------------------------------------------
// Fused local attention (window=1024, look_backward=1), one wave per 16-row Q
// tile of one (b, h, window). Online softmax; WMMA for QK^T and P*V.
// V chunks staged global->LDS with the async DMA path (ASYNCcnt).
// Q/K/V: (b, h, 2048, 64) f16.  O: (b, 2048, 512) f16 (heads concatenated).
// ---------------------------------------------------------------------------
__global__ __launch_bounds__(32) void k_local_attn(
    const h16* __restrict__ Q, const h16* __restrict__ K,
    const h16* __restrict__ V, h16* __restrict__ O)
{
  __shared__ __align__(32) h16 Pbuf[16 * 32];   // P tile, row-major [m][k]
  __shared__ __align__(32) h16 Vrow[32 * 64];   // V chunk, row-major [key][feat]
  const int lane = threadIdx.x;
  const bool hi  = lane >= 16;
  const int ln   = lane & 15;
  const int bid  = blockIdx.x;
  const int tile = bid & 63;
  const int win  = (bid >> 6) & 1;
  const int hd   = (bid >> 7) & 7;
  const int bb   = bid >> 10;
  const int t0   = tile * 16;
  const int bh   = bb * 8 + hd;

  const h16* qrow = Q + ((size_t)(bh * 2048 + win * 1024 + t0 + ln)) * 64;
  v16h qa0 = ld_a16(qrow, 0, hi);
  v16h qa1 = ld_a16(qrow, 32, hi);

  // LDS byte offset of Vrow (flat-address low 32 bits == LDS offset)
  const unsigned vrow_lds = (unsigned)(size_t)(&Vrow[0]) + (unsigned)(lane * 128);

  float mrow[8], lrow[8], fac[8];
  v8f oacc[4];
#pragma unroll
  for (int r = 0; r < 8; ++r) { mrow[r] = -3.0e38f; lrow[r] = 0.f; }
#pragma unroll
  for (int c = 0; c < 4; ++c) oacc[c] = v8zero();

  const int jstart = (win == 0) ? 1024 : 0;
  int jend = (t0 + 1024 + 16 + 31) & ~31;   // causal upper bound, chunk-aligned
  if (jend > 2048) jend = 2048;
  const int krow0 = bh * 2048 + (win - 1) * 1024;  // kk[j] = K row (win-1)*W + j

  for (int j = jstart; j < jend; j += 32) {
    // --- async DMA: stage V chunk rows (32 x 128B) into LDS ---------------
    {
      const h16* vg = V + ((size_t)(krow0 + j + lane)) * 64;  // lane = key row
#pragma unroll
      for (int s8 = 0; s8 < 8; ++s8) {
        unsigned loff = vrow_lds + (unsigned)(s8 * 16);
        const h16* gp = vg + s8 * 8;
        asm volatile("global_load_async_to_lds_b128 %0, %1, off"
                     :: "v"(loff), "v"(gp) : "memory");
      }
    }
    // --- scores: sim = Q @ Ktile^T  (K rows load directly as B operands) ---
    v8f s0 = v8zero(), s1 = v8zero();
    {
      const h16* kp0 = K + ((size_t)(krow0 + j + ln)) * 64 + (hi ? 16 : 0);
      s0 = wmma16(qa0, ld_b16(kp0), s0);
      s0 = wmma16(qa1, ld_b16(kp0 + 32), s0);
      const h16* kp1 = K + ((size_t)(krow0 + j + 16 + ln)) * 64 + (hi ? 16 : 0);
      s1 = wmma16(qa0, ld_b16(kp1), s1);
      s1 = wmma16(qa1, ld_b16(kp1 + 32), s1);
    }
    const int jk0 = j + ln, jk1 = j + 16 + ln;
    // --- scale, mask, online softmax (row reductions across 16-lane halves) -
#pragma unroll
    for (int r = 0; r < 8; ++r) {
      const int iq = t0 + r + (hi ? 8 : 0);
      float v0 = s0[r] * 0.125f;   // DH^-0.5
      float v1 = s1[r] * 0.125f;
      if (!((jk0 <= iq + 1024) && (win > 0 || jk0 >= 1024))) v0 = -1.0e30f;
      if (!((jk1 <= iq + 1024) && (win > 0 || jk1 >= 1024))) v1 = -1.0e30f;
      float mx = fmaxf(v0, v1);
#pragma unroll
      for (int d = 8; d >= 1; d >>= 1) mx = fmaxf(mx, __shfl_xor(mx, d, 32));
      const float mnew = fmaxf(mrow[r], mx);
      fac[r] = __expf(mrow[r] - mnew);
      mrow[r] = mnew;
      const float p0 = __expf(v0 - mnew);
      const float p1 = __expf(v1 - mnew);
      float rs = p0 + p1;
#pragma unroll
      for (int d = 8; d >= 1; d >>= 1) rs += __shfl_xor(rs, d, 32);
      lrow[r] = lrow[r] * fac[r] + rs;
      Pbuf[(r + (hi ? 8 : 0)) * 32 + ln]      = (h16)p0;
      Pbuf[(r + (hi ? 8 : 0)) * 32 + 16 + ln] = (h16)p1;
    }
#pragma unroll
    for (int c = 0; c < 4; ++c)
#pragma unroll
      for (int r = 0; r < 8; ++r) oacc[c][r] *= fac[r];
    // --- wait for DMA, then oacc += P @ Vchunk ----------------------------
    asm volatile("s_wait_asynccnt 0" ::: "memory");
    __syncthreads();
    v16h pa = ld_a16(&Pbuf[ln * 32], 0, hi);
    const int kb = hi ? 16 : 0;   // B operand key range for this lane half
#pragma unroll
    for (int c = 0; c < 4; ++c) {
      V16U u;
      const int n = c * 16 + ln;
#pragma unroll
      for (int e = 0; e < 16; ++e) u.e[e] = Vrow[(kb + e) * 64 + n];
      oacc[c] = wmma16(pa, u.v, oacc[c]);
    }
    __syncthreads();   // ds reads done before next chunk's DMA overwrites LDS
  }
  // --- normalize and store into (b, n, h*64) ------------------------------
#pragma unroll
  for (int c = 0; c < 4; ++c)
#pragma unroll
    for (int r = 0; r < 8; ++r) {
      const int seq = win * 1024 + t0 + r + (hi ? 8 : 0);
      O[((size_t)(bb * 2048 + seq)) * 512 + hd * 64 + c * 16 + ln] =
          (h16)(oacc[c][r] / lrow[r]);
    }
}

// --------------------------- small VALU kernels -----------------------------

__global__ __launch_bounds__(128) void k_layernorm(
    const float* __restrict__ X, const float* __restrict__ G,
    const float* __restrict__ Bv, h16* __restrict__ Ot, int C)
{
  const int row  = blockIdx.x * 4 + (threadIdx.x >> 5);
  const int lane = threadIdx.x & 31;
  const float* xr = X + (size_t)row * C;
  float s = 0.f, s2 = 0.f;
  for (int i = lane; i < C; i += 32) { float v = xr[i]; s += v; s2 += v * v; }
#pragma unroll
  for (int d = 16; d >= 1; d >>= 1) {
    s += __shfl_xor(s, d, 32);
    s2 += __shfl_xor(s2, d, 32);
  }
  const float mean = s / C;
  const float var  = s2 / C - mean * mean;
  const float rinv = rsqrtf(var + 1e-5f);
  h16* orow = Ot + (size_t)row * C;
  for (int i = lane; i < C; i += 32)
    orow[i] = (h16)((xr[i] - mean) * rinv * G[i] + Bv[i]);
}

// qkv (b,n,1536) f32 -> rotary q/k + v, each (b,h,2048,64) f16
__global__ void k_rotary(const float* __restrict__ qkv,
                         h16* __restrict__ q16, h16* __restrict__ k16,
                         h16* __restrict__ v16o)
{
  const int idx = blockIdx.x * blockDim.x + threadIdx.x; // b*h*n*32 threads
  const int i  = idx & 31;
  const int n  = (idx >> 5) & 2047;
  const int hh = (idx >> 16) & 7;
  const int b  = idx >> 19;
  const float inv = powf(10000.f, -(float)i * (1.f / 32.f));
  const float f  = (float)n * inv;
  const float cs = cosf(f), sn = sinf(f);
  const size_t base = ((size_t)(b * 2048 + n)) * 1536 + hh * 64;
  const float q1 = qkv[base + i],        q2 = qkv[base + 32 + i];
  const float k1 = qkv[base + 512 + i],  k2 = qkv[base + 512 + 32 + i];
  const float v1 = qkv[base + 1024 + i], v2 = qkv[base + 1024 + 32 + i];
  const size_t ob = ((size_t)((b * 8 + hh) * 2048 + n)) * 64;
  q16[ob + i]      = (h16)(q1 * cs - q2 * sn);
  q16[ob + 32 + i] = (h16)(q2 * cs + q1 * sn);
  k16[ob + i]      = (h16)(k1 * cs - k2 * sn);
  k16[ob + 32 + i] = (h16)(k2 * cs + k1 * sn);
  v16o[ob + i]      = (h16)v1;
  v16o[ob + 32 + i] = (h16)v2;
}

__global__ void k_f32_to_f16(const float* __restrict__ s, h16* __restrict__ d,
                             long long n)
{
  long long i = (long long)blockIdx.x * blockDim.x + threadIdx.x;
  if (i < n) d[i] = (h16)s[i];
}

__global__ void k_zero16(h16* __restrict__ d, int n)
{
  int i = blockIdx.x * blockDim.x + threadIdx.x;
  if (i < n) d[i] = (h16)0.f;
}

// conv_out: 32->1, k=7 causal. X: (2, L, 32) f16, W: (1,32,7) f32.
__global__ void k_conv_out(const h16* __restrict__ X, const float* __restrict__ Wc,
                           const float* __restrict__ Bc, float* __restrict__ Y,
                           int L)
{
  long long i = (long long)blockIdx.x * blockDim.x + threadIdx.x;
  if (i >= 2LL * L) return;
  const int bz = (int)(i / L);
  const int l  = (int)(i % L);
  float acc = Bc[0];
  for (int t = 0; t < 7; ++t) {
    const int ll = l + t - 6;
    if (ll < 0) continue;
    const h16* xr = X + ((size_t)bz * L + ll) * 32;
    for (int ci = 0; ci < 32; ++ci) acc += (float)xr[ci] * Wc[ci * 7 + t];
  }
  Y[i] = acc;
}

// ------------------------ weight preparation kernels ------------------------
// Plain weight (K,N) f32 -> Bt (N,K) f16
__global__ void k_prep_mat(const float* __restrict__ src, h16* __restrict__ dst,
                           int K, int N)
{
  int i = blockIdx.x * blockDim.x + threadIdx.x;
  if (i >= K * N) return;
  int k = i / N, n = i % N;
  dst[(size_t)n * K + k] = (h16)src[i];
}
// Conv weight (Co,Ci,T) f32 -> (T,Co,Ci) f16  (Bt per tap)
__global__ void k_prep_conv(const float* __restrict__ src, h16* __restrict__ dst,
                            int Co, int Ci, int T)
{
  int i = blockIdx.x * blockDim.x + threadIdx.x;
  if (i >= Co * Ci * T) return;
  int co = i / (Ci * T);
  int r  = i % (Ci * T);
  int ci = r / T, t = r % T;
  dst[((size_t)t * Co + co) * Ci + ci] = (h16)src[i];
}
// ConvT weight (Ci,Co,K) f32 -> (K,Co,Ci) f16 (Bt per tap)
__global__ void k_prep_convT(const float* __restrict__ src, h16* __restrict__ dst,
                             int Ci, int Co, int Kk)
{
  int i = blockIdx.x * blockDim.x + threadIdx.x;
  if (i >= Ci * Co * Kk) return;
  int ci = i / (Co * Kk);
  int r  = i % (Co * Kk);
  int co = r / Kk, kk = r % Kk;
  dst[((size_t)kk * Co + co) * Ci + ci] = (h16)src[i];
}

// ============================================================================
// Host orchestration
// ============================================================================
extern "C" void kernel_launch(void* const* d_in, const int* in_sizes, int n_in,
                              void* d_out, int out_size, void* d_ws, size_t ws_size,
                              hipStream_t stream)
{
  (void)in_sizes; (void)n_in; (void)out_size; (void)ws_size;
  const int B = 2, N = 2048, C = 512, HD = 8, DH = 64;

  char* base = (char*)d_ws;
  size_t cur = 0;
  auto alloc = [&](size_t bytes) -> char* {
    cur = (cur + 255) & ~(size_t)255;
    char* p = base + cur;
    cur += bytes;
    return p;
  };

  // ---- activation buffers ----
  float* qkv = (float*)alloc((size_t)B * N * 1536 * 4);
  float* xA  = (float*)alloc((size_t)B * N * C * 4);
  float* xB  = (float*)alloc((size_t)B * N * C * 4);
  h16* hln = (h16*)alloc((size_t)B * N * C * 2);
  h16* q16 = (h16*)alloc((size_t)B * HD * N * DH * 2);
  h16* k16 = (h16*)alloc((size_t)B * HD * N * DH * 2);
  h16* v16 = (h16*)alloc((size_t)B * HD * N * DH * 2);
  h16* o16 = (h16*)alloc((size_t)B * N * C * 2);
  h16* h1  = (h16*)alloc((size_t)B * N * 2048 * 2);
  h16* zrow = (h16*)alloc((size_t)2048 * 2);          // zeroed dummy A row
  const size_t DBYTES = (size_t)2 * 524288 * 32 * 2;  // 64MB: max decoder act
  h16* Dbuf[3] = { (h16*)alloc(DBYTES), (h16*)alloc(DBYTES), (h16*)alloc(DBYTES) };

  k_zero16<<<(2048 + 255) / 256, 256, 0, stream>>>(zrow, 2048);

  // ---- weight prep (f32 -> transposed f16), idempotent per call ----
  auto prepMat = [&](const void* src, int K, int Nn) -> h16* {
    h16* dst = (h16*)alloc((size_t)K * Nn * 2);
    int tot = K * Nn;
    k_prep_mat<<<(tot + 255) / 256, 256, 0, stream>>>((const float*)src, dst, K, Nn);
    return dst;
  };
  auto prepConv = [&](const void* src, int Co, int Ci, int T) -> h16* {
    h16* dst = (h16*)alloc((size_t)Co * Ci * T * 2);
    int tot = Co * Ci * T;
    k_prep_conv<<<(tot + 255) / 256, 256, 0, stream>>>((const float*)src, dst, Co, Ci, T);
    return dst;
  };
  auto prepConvT = [&](const void* src, int Ci, int Co, int Kk) -> h16* {
    h16* dst = (h16*)alloc((size_t)Ci * Co * Kk * 2);
    int tot = Ci * Co * Kk;
    k_prep_convT<<<(tot + 255) / 256, 256, 0, stream>>>((const float*)src, dst, Ci, Co, Kk);
    return dst;
  };

  // input order: x, then attn layers {ln1_g,ln1_b,wqkv,wo,ln2_g,ln2_b,w1,b1,w2,b2},
  // conv_in {w,b}, blocks {ct_w,ct_b, 3x{w1,b1,w2,b2}}, conv_out {w,b}
  h16 *wqkvT[2], *woT[2], *w1T[2], *w2T[2];
  const float *ln1g[2], *ln1b[2], *ln2g[2], *ln2b[2], *b1p[2], *b2p[2];
  for (int l = 0; l < 2; ++l) {
    int bi = 1 + l * 10;
    ln1g[l] = (const float*)d_in[bi + 0];
    ln1b[l] = (const float*)d_in[bi + 1];
    wqkvT[l] = prepMat(d_in[bi + 2], 512, 1536);
    woT[l]   = prepMat(d_in[bi + 3], 512, 512);
    ln2g[l] = (const float*)d_in[bi + 4];
    ln2b[l] = (const float*)d_in[bi + 5];
    w1T[l] = prepMat(d_in[bi + 6], 512, 2048);
    b1p[l] = (const float*)d_in[bi + 7];
    w2T[l] = prepMat(d_in[bi + 8], 2048, 512);
    b2p[l] = (const float*)d_in[bi + 9];
  }
  h16* ciW = prepConv(d_in[21], 512, 512, 7);
  const float* ciB = (const float*)d_in[22];
  const int chans[5] = {512, 256, 128, 64, 32};
  const int strid[4] = {8, 4, 4, 2};
  const int dils[3]  = {1, 3, 9};
  h16 *ctW[4], *rw1[4][3], *rw2[4][3];
  const float *ctB[4], *rb1[4][3], *rb2[4][3];
  for (int i = 0; i < 4; ++i) {
    int bi = 23 + i * 14;
    int ci = chans[i], co = chans[i + 1], kk = 2 * strid[i];
    ctW[i] = prepConvT(d_in[bi], ci, co, kk);
    ctB[i] = (const float*)d_in[bi + 1];
    for (int r = 0; r < 3; ++r) {
      int rb = bi + 2 + r * 4;
      rw1[i][r] = prepConv(d_in[rb], co, co, 7);
      rb1[i][r] = (const float*)d_in[rb + 1];
      rw2[i][r] = prepConv(d_in[rb + 2], co, co, 1);
      rb2[i][r] = (const float*)d_in[rb + 3];
    }
  }

  auto gemm = [&](const h16* Aa, const h16* Ww, const float* bb,
                  const h16* Rh, const float* Rf, h16* Yh, float* Yf,
                  int M, int LA, int Cin, int Cout, int ntaps, int dr,
                  int w0, int wstep, int ostride, int ophase,
                  long long sAB, long long sYB, int act) {
    dim3 g((Cout + 63) / 64, M / 64, B);
    k_gemm_taps<<<g, 128, 0, stream>>>(Aa, Ww, bb, Rh, Rf, Yh, Yf, zrow, M, LA,
                                       Cin, Cout, ntaps, dr, w0, wstep, ostride,
                                       ophase, sAB, sYB, act);
  };

  // ---- transformer (residual stream f32, GEMM operands f16) ----
  const float* xc = (const float*)d_in[0];
  for (int l = 0; l < 2; ++l) {
    k_layernorm<<<B * N / 4, 128, 0, stream>>>(xc, ln1g[l], ln1b[l], hln, C);
    gemm(hln, wqkvT[l], nullptr, nullptr, nullptr, nullptr, qkv,
         N, N, 512, 1536, 1, 0, 0, 0, 1, 0, (long long)N * 512, (long long)N * 1536, 0);
    k_rotary<<<(B * HD * N * 32) / 256, 256, 0, stream>>>(qkv, q16, k16, v16);
    k_local_attn<<<2048, 32, 0, stream>>>(q16, k16, v16, o16);
    gemm(o16, woT[l], nullptr, nullptr, xc, nullptr, xA,           // x += o@wo
         N, N, 512, 512, 1, 0, 0, 0, 1, 0, (long long)N * 512, (long long)N * 512, 0);
    k_layernorm<<<B * N / 4, 128, 0, stream>>>(xA, ln2g[l], ln2b[l], hln, C);
    gemm(hln, w1T[l], b1p[l], nullptr, nullptr, h1, nullptr,       // gelu(h@w1+b1)
         N, N, 512, 2048, 1, 0, 0, 0, 1, 0, (long long)N * 512, (long long)N * 2048, 1);
    gemm(h1, w2T[l], b2p[l], nullptr, xA, nullptr, xB,             // x += h1@w2+b2
         N, N, 2048, 512, 1, 0, 0, 0, 1, 0, (long long)N * 2048, (long long)N * 512, 0);
    xc = xB;
  }

  // ---- decoder ((b, l, c) f16 activations) ----
  k_f32_to_f16<<<(B * N * C + 255) / 256, 256, 0, stream>>>(xc, Dbuf[0],
                                                            (long long)B * N * C);
  gemm(Dbuf[0], ciW, ciB, nullptr, nullptr, Dbuf[1], nullptr,      // conv_in k7
       N, N, 512, 512, 7, 1, 0, 1, 1, 0, (long long)N * 512, (long long)N * 512, 0);
  int xi = 1;
  int Lc = N;
  for (int i = 0; i < 4; ++i) {
    const int s = strid[i], ci = chans[i], co = chans[i + 1];
    const int nb = (xi + 1) % 3;
    for (int p = 0; p < s; ++p)   // convT phase p: y[m*s+p] = x[m]@W[p] + x[m-1]@W[p+s]
      gemm(Dbuf[xi], ctW[i], ctB[i], nullptr, nullptr, Dbuf[nb], nullptr,
           Lc, Lc, ci, co, 2, 1, p + s, -s, s, p,
           (long long)Lc * ci, (long long)Lc * s * co, 0);
    xi = nb;
    Lc *= s;
    for (int r = 0; r < 3; ++r) {
      const int hb = (xi + 1) % 3, ob = (xi + 2) % 3;
      gemm(Dbuf[xi], rw1[i][r], rb1[i][r], nullptr, nullptr, Dbuf[hb], nullptr,
           Lc, Lc, co, co, 7, dils[r], 0, 1, 1, 0,                 // elu(conv k7 dil)
           (long long)Lc * co, (long long)Lc * co, 2);
      gemm(Dbuf[hb], rw2[i][r], rb2[i][r], Dbuf[xi], nullptr, Dbuf[ob], nullptr,
           Lc, Lc, co, co, 1, 0, 0, 0, 1, 0,                       // x + elu(conv1x1)
           (long long)Lc * co, (long long)Lc * co, 2);
      xi = ob;
    }
  }
  k_conv_out<<<(2 * 524288 + 255) / 256, 256, 0, stream>>>(
      Dbuf[xi], (const float*)d_in[79], (const float*)d_in[80], (float*)d_out,
      524288);
}
